// SelfAttentionBlock_69243462746494
// MI455X (gfx1250) — compile-verified, matrix operand order checked by reading, not run on hardware
//
#include <hip/hip_runtime.h>

// ---------------------------------------------------------------------------
// Self-attention block (QKV proj -> MHA -> out proj -> residual -> GroupNorm)
// CDNA5 / gfx1250: GEMMs on v_wmma_f32_16x16x32_bf16 fed by TDM
// (tensor_load_to_lds, TENSORcnt) + async-to-LDS (ASYNCcnt) double-buffered
// staging; WMMA flash attention; wave32.
// ---------------------------------------------------------------------------

typedef __attribute__((ext_vector_type(16))) __bf16 v16bf;
typedef __attribute__((ext_vector_type(8)))  float  v8f;
typedef unsigned int u32x4 __attribute__((ext_vector_type(4)));
typedef int          i32x8 __attribute__((ext_vector_type(8)));
typedef int          i32x4 __attribute__((ext_vector_type(4)));

#define DEVFN __device__ __forceinline__

#if __has_builtin(__builtin_amdgcn_tensor_load_to_lds)
#define HAVE_TDM 1
#else
#define HAVE_TDM 0
#endif

namespace cfg {
constexpr int B  = 8;
constexpr int C  = 512;          // input channels
constexpr int E  = 512;          // embed dim
constexpr int H  = 32, W = 32;
constexpr int S  = H * W;        // 1024 tokens
constexpr int NH = 8;            // heads
constexpr int HD = E / NH;       // 64 head dim
constexpr int NG = 32;           // groupnorm groups
constexpr int CPG = E / NG;      // 16 channels per group
constexpr float QSCALE = 0.125f; // hd^-0.5
constexpr float EPS = 1e-5f;
}

union FragU { uint4 u[2]; v16bf v; };

// Load a 16-lane-striped WMMA bf16 fragment from a row-major [16][>=32] tile.
// ISA layout (16-bit A 16x32 / B 32x16): lane l -> row = l&15, half = l>>4,
// halves K in [h*8, h*8+8) and [16+h*8, 16+h*8+8): two contiguous b128 loads.
DEVFN v16bf ldfrag(const __bf16* base, int strideHalves, int lane) {
  const int r = lane & 15;
  const int h = (lane >> 4) & 1;
  const __bf16* p = base + (long)r * strideHalves + h * 8;
  FragU f;
  f.u[0] = *reinterpret_cast<const uint4*>(p);
  f.u[1] = *reinterpret_cast<const uint4*>(p + 16);
  return f.v;
}

DEVFN v8f wmma_bf16(v16bf a, v16bf b, v8f c) {
  // D = A(16x32) * B(32x16) + C, f32 accumulate
  return __builtin_amdgcn_wmma_f32_16x16x32_bf16(false, a, false, b,
                                                 (short)0, c, false, false);
}

// ---------------------------------------------------------------------------
// TDM: one descriptor-driven DMA of a 2D tile (128 rows x 32 bf16, row
// stride `strideHalves` elements) from global to LDS. Issued by one wave;
// tracked by TENSORcnt. D# bit layout per CDNA5 ISA ch.8.
// ---------------------------------------------------------------------------
DEVFN void tdm_stage_128x32(const __bf16* src, int strideHalves, unsigned ldsoff) {
#if HAVE_TDM
  const unsigned long ga = (unsigned long)src;
  u32x4 g0;
  g0[0] = 1u;                                  // count=1 (valid user D#)
  g0[1] = ldsoff;                              // lds_addr (bytes)
  g0[2] = (unsigned)ga;                        // global_addr[31:0]
  g0[3] = (unsigned)(ga >> 32) | (2u << 30);   // global_addr[56:32] | type=2
  i32x8 g1;
  g1[0] = 1 << 16;        // workgroup_mask=0 | data_size=1 (2 bytes)
  g1[1] = 32 << 16;       // tensor_dim0[15:0]=32 (| atomic_barrier_addr=0)
  g1[2] = 128 << 16;      // tensor_dim1[15:0]=128 | tensor_dim0[31:16]=0
  g1[3] = 32 << 16;       // tile_dim0=32 | tensor_dim1[31:16]=0
  g1[4] = 128;            // tile_dim2=0 | tile_dim1=128
  g1[5] = strideHalves;   // tensor_dim0_stride[31:0]
  g1[6] = 0;              // tensor_dim1_stride[15:0] | stride0[47:32]
  g1[7] = 0;              // tensor_dim1_stride[47:16]
  const i32x4 z4 = {0, 0, 0, 0};               // groups 2/3: dims 3/4 unused
  const i32x8 z8 = {0, 0, 0, 0, 0, 0, 0, 0};   // trailing group (6-arg form)
  __builtin_amdgcn_tensor_load_to_lds(g0, g1, z4, z4, z8, 0);
#else
  (void)src; (void)strideHalves; (void)ldsoff;
#endif
}

// ---------------------------------------------------------------------------
// Async (ASYNCcnt) global -> LDS staging of a 128-row x 32-half bf16 tile.
// 256 threads x 2 chunks x 16B = 8KB; per-lane 64-bit gaddr + 32-bit LDS addr.
// ---------------------------------------------------------------------------
DEVFN void stage_async_tile(const __bf16* gbase, int rowStride, int c0,
                            __bf16 (*lds)[32], int tid) {
#pragma unroll
  for (int q = 0; q < 2; ++q) {
    const int ci = tid * 2 + q;          // 0..511
    const int r  = ci >> 2;              // 0..127
    const int cp = (ci & 3) * 8;         // 0,8,16,24 halves
    const __bf16* src = gbase + (long)r * rowStride + (c0 + cp);
    const unsigned ldsoff = (unsigned)(uintptr_t)(&lds[r][cp]);
    asm volatile("global_load_async_to_lds_b128 %0, %1, off"
                 :: "v"(ldsoff), "v"(src)
                 : "memory");
  }
}

// Stage one K-step: A tile via TDM (wave 0), B tile via async lane loads.
DEVFN void stage_all(const __bf16* Ab, const __bf16* Bb, int KS, int c0,
                     __bf16 (*sa)[32], __bf16 (*sb)[32], int tid, int w) {
#if HAVE_TDM
  if (w == 0)
    tdm_stage_128x32(Ab + c0, KS, (unsigned)(uintptr_t)(&sa[0][0]));
  stage_async_tile(Bb, KS, c0, sb, tid);
#else
  stage_async_tile(Ab, KS, c0, sa, tid);
  stage_async_tile(Bb, KS, c0, sb, tid);
#endif
}

// Drain previous stage (ops complete in order per counter): after issuing the
// next stage, waiting "<= ops-just-issued" guarantees the current buffer.
DEVFN void wait_stage(bool more, int w) {
#if HAVE_TDM
  if (w == 0) {
    if (more) __builtin_amdgcn_s_wait_tensorcnt(1);
    else      __builtin_amdgcn_s_wait_tensorcnt(0);
  }
  if (more) asm volatile("s_wait_asynccnt 0x2" ::: "memory");
  else      asm volatile("s_wait_asynccnt 0x0" ::: "memory");
#else
  if (more) asm volatile("s_wait_asynccnt 0x4" ::: "memory");
  else      asm volatile("s_wait_asynccnt 0x0" ::: "memory");
#endif
}

// ---------------------------------------------------------------------------
// Shared GEMM core: block tile 128(M) x 128(N), K-step 32, double-buffered
// TDM+async LDS staging. 8 waves as 4(M) x 2(N); wave tile 32 x 64 (8 accums).
// Ab: row-major [M][KS] at block M-base; Bb: row-major [N][KS] at block N-base.
// ---------------------------------------------------------------------------
DEVFN void gemm_core_128x128(const __bf16* __restrict__ Ab,
                             const __bf16* __restrict__ Bb, int KS,
                             __bf16 (*sA)[128][32], __bf16 (*sB)[128][32],
                             int lane, int w, int tid, v8f acc[2][4]) {
  const int wm = (w >> 1) * 32;
  const int wn = (w & 1) * 64;
  stage_all(Ab, Bb, KS, 0, sA[0], sB[0], tid, w);
  int buf = 0;
  for (int c0 = 0; c0 < KS; c0 += 32) {
    const bool more = (c0 + 32 < KS);
    if (more)
      stage_all(Ab, Bb, KS, c0 + 32, sA[buf ^ 1], sB[buf ^ 1], tid, w);
    wait_stage(more, w);
    __syncthreads();      // all staging of `buf` visible to every wave

    v16bf a0 = ldfrag(&sA[buf][wm +  0][0], 32, lane);
    v16bf a1 = ldfrag(&sA[buf][wm + 16][0], 32, lane);
    v16bf b0 = ldfrag(&sB[buf][wn +  0][0], 32, lane);
    v16bf b1 = ldfrag(&sB[buf][wn + 16][0], 32, lane);
    v16bf b2 = ldfrag(&sB[buf][wn + 32][0], 32, lane);
    v16bf b3 = ldfrag(&sB[buf][wn + 48][0], 32, lane);
    acc[0][0] = wmma_bf16(a0, b0, acc[0][0]);
    acc[0][1] = wmma_bf16(a0, b1, acc[0][1]);
    acc[0][2] = wmma_bf16(a0, b2, acc[0][2]);
    acc[0][3] = wmma_bf16(a0, b3, acc[0][3]);
    acc[1][0] = wmma_bf16(a1, b0, acc[1][0]);
    acc[1][1] = wmma_bf16(a1, b1, acc[1][1]);
    acc[1][2] = wmma_bf16(a1, b2, acc[1][2]);
    acc[1][3] = wmma_bf16(a1, b3, acc[1][3]);

    __syncthreads();      // nobody may overwrite `buf` until all read it
    buf ^= 1;
  }
}

// ---------------------------------------------------------------------------
// Transpose + convert: f32 [rows][cols] -> bf16 [cols][rows]
// ---------------------------------------------------------------------------
__global__ void __launch_bounds__(256) k_transpose_f32_bf16(
    const float* __restrict__ in, __bf16* __restrict__ out,
    int rows, int cols, long inBatch, long outBatch) {
  __shared__ float tile[32][33];
  const float* src = in + (long)blockIdx.z * inBatch;
  __bf16* dst = out + (long)blockIdx.z * outBatch;
  const int c0 = blockIdx.x * 32, r0 = blockIdx.y * 32;
  const int tx = threadIdx.x & 31, ty = threadIdx.x >> 5;
#pragma unroll
  for (int k = 0; k < 32; k += 8)
    tile[ty + k][tx] = src[(long)(r0 + ty + k) * cols + (c0 + tx)];
  __syncthreads();
#pragma unroll
  for (int k = 0; k < 32; k += 8)
    dst[(long)(c0 + ty + k) * rows + (r0 + tx)] = (__bf16)tile[tx][ty + k];
}

// bf16 [rows][cols] -> bf16 [cols][rows]  (used for V -> V^T per head)
__global__ void __launch_bounds__(256) k_transpose_bf16(
    const __bf16* __restrict__ in, __bf16* __restrict__ out,
    int rows, int cols, long inBatch, long outBatch) {
  __shared__ __bf16 tile[32][33];
  const __bf16* src = in + (long)blockIdx.z * inBatch;
  __bf16* dst = out + (long)blockIdx.z * outBatch;
  const int c0 = blockIdx.x * 32, r0 = blockIdx.y * 32;
  const int tx = threadIdx.x & 31, ty = threadIdx.x >> 5;
#pragma unroll
  for (int k = 0; k < 32; k += 8)
    tile[ty + k][tx] = src[(long)(r0 + ty + k) * cols + (c0 + tx)];
  __syncthreads();
#pragma unroll
  for (int k = 0; k < 32; k += 8)
    dst[(long)(c0 + ty + k) * rows + (r0 + tx)] = tile[tx][ty + k];
}

// ---------------------------------------------------------------------------
// QKV projection: out[b,h,s,d] = (xb[b,s,:] @ W + bias) (* QSCALE for Q)
// xb  : bf16 [B][S][C];  Wts : bf16 [3][E][C] (pre-transposed)
// ---------------------------------------------------------------------------
__global__ void __launch_bounds__(256) k_gemm_qkv(
    const __bf16* __restrict__ xb, const __bf16* __restrict__ Wts,
    const float* __restrict__ bq, const float* __restrict__ bk,
    const float* __restrict__ bv,
    __bf16* __restrict__ Qo, __bf16* __restrict__ Ko, __bf16* __restrict__ Vo) {
  using namespace cfg;
  __shared__ __bf16 sA[2][128][32];
  __shared__ __bf16 sB[2][128][32];

  const int z = blockIdx.z;
  const int b = z / 3, mat = z % 3;
  const __bf16* Wt = Wts + (long)mat * E * C;
  const float* bias = (mat == 0) ? bq : (mat == 1) ? bk : bv;
  __bf16* outp = (mat == 0) ? Qo : (mat == 1) ? Ko : Vo;
  const float oscale = (mat == 0) ? QSCALE : 1.0f;

  const int tid = threadIdx.x;
  const int lane = tid & 31, w = tid >> 5;
  const int sBlk = blockIdx.y * 128, eBlk = blockIdx.x * 128;

  v8f acc[2][4] = {};
  gemm_core_128x128(xb + ((long)b * S + sBlk) * C, Wt + (long)eBlk * C, C,
                    sA, sB, lane, w, tid, acc);

  const int n = lane & 15, hs = lane >> 4;
  const int s0w = sBlk + (w >> 1) * 32;
  const int e0w = eBlk + (w & 1) * 64;
#pragma unroll
  for (int mi = 0; mi < 2; ++mi) {
#pragma unroll
    for (int ni = 0; ni < 4; ++ni) {
      const int e = e0w + ni * 16 + n;
      const int h = e >> 6, d = e & (HD - 1);
      const float bb = bias[e];
      __bf16* op = outp + (((long)b * NH + h) * S) * HD + d;
#pragma unroll
      for (int j = 0; j < 8; ++j) {
        const int s = s0w + mi * 16 + hs * 8 + j;
        op[(long)s * HD] = (__bf16)((acc[mi][ni][j] + bb) * oscale);
      }
    }
  }
}

// ---------------------------------------------------------------------------
// Flash attention: per (b,h), wave owns 16 q-rows; online softmax over 32-wide
// key blocks. Q pre-scaled. Output attn -> bf16 [B][S][E] (e = h*64+d).
// ---------------------------------------------------------------------------
__global__ void __launch_bounds__(256) k_attention(
    const __bf16* __restrict__ Q, const __bf16* __restrict__ K,
    const __bf16* __restrict__ Vt,   // [B*NH][HD][S]
    __bf16* __restrict__ attnb) {    // [B][S][E]
  using namespace cfg;
  __shared__ __bf16 sP[8][16][32];   // per-wave P staging (C-layout -> A-layout)

  const int bh = blockIdx.y;
  const int b = bh >> 3, h = bh & 7;
  const int lane = threadIdx.x & 31, w = threadIdx.x >> 5;
  const int q0 = blockIdx.x * 128 + w * 16;

  const __bf16* Qp = Q + ((long)bh * S + q0) * HD;
  const __bf16* Kp = K + (long)bh * S * HD;
  const __bf16* Vp = Vt + (long)bh * HD * S;

  const v16bf aQ0 = ldfrag(Qp, HD, lane);
  const v16bf aQ1 = ldfrag(Qp + 32, HD, lane);

  v8f o0 = {}, o1 = {}, o2 = {}, o3 = {};
  float m[8], l_[8];
#pragma unroll
  for (int j = 0; j < 8; ++j) { m[j] = -1e30f; l_[j] = 0.0f; }

  __bf16 (*myP)[32] = sP[w];
  const int n = lane & 15, hs = lane >> 4;

  for (int kc = 0; kc < S; kc += 32) {
    // scores = Q * K^T  (16 q x 32 k), contraction over d=64
    v16bf bK00 = ldfrag(Kp + (long)kc * HD, HD, lane);
    v16bf bK01 = ldfrag(Kp + (long)kc * HD + 32, HD, lane);
    v16bf bK10 = ldfrag(Kp + (long)(kc + 16) * HD, HD, lane);
    v16bf bK11 = ldfrag(Kp + (long)(kc + 16) * HD + 32, HD, lane);
    v8f s0v = {}, s1v = {};
    s0v = wmma_bf16(aQ0, bK00, s0v);
    s0v = wmma_bf16(aQ1, bK01, s0v);
    s1v = wmma_bf16(aQ0, bK10, s1v);
    s1v = wmma_bf16(aQ1, bK11, s1v);

    // online softmax; a row's 16 values live in one 16-lane half (C layout)
    float corr[8];
#pragma unroll
    for (int j = 0; j < 8; ++j) {
      float mx = fmaxf(s0v[j], s1v[j]);
#pragma unroll
      for (int msk = 1; msk <= 8; msk <<= 1)
        mx = fmaxf(mx, __shfl_xor(mx, msk, 32));
      const float mn = fmaxf(m[j], mx);
      corr[j] = __expf(m[j] - mn);
      m[j] = mn;
      const float p0 = __expf(s0v[j] - mn);
      const float p1 = __expf(s1v[j] - mn);
      s0v[j] = p0; s1v[j] = p1;
      float rs = p0 + p1;
#pragma unroll
      for (int msk = 1; msk <= 8; msk <<= 1)
        rs += __shfl_xor(rs, msk, 32);
      l_[j] = l_[j] * corr[j] + rs;
    }
#pragma unroll
    for (int j = 0; j < 8; ++j) {
      o0[j] *= corr[j]; o1[j] *= corr[j];
      o2[j] *= corr[j]; o3[j] *= corr[j];
    }

    // stage P (16x32 bf16) row-major in wave-private LDS, reload as A fragment
#pragma unroll
    for (int j = 0; j < 8; ++j) {
      const int mr = hs * 8 + j;
      myP[mr][n]      = (__bf16)s0v[j];
      myP[mr][16 + n] = (__bf16)s1v[j];
    }
    asm volatile("s_wait_dscnt 0" ::: "memory");  // wave-local LDS RAW fence
    v16bf aP = ldfrag(&myP[0][0], 32, lane);

    // O += P * V   (V^T rows are S-contiguous -> direct B fragments)
    v16bf bV0 = ldfrag(Vp + 0L * 16 * S + kc, S, lane);
    v16bf bV1 = ldfrag(Vp + 1L * 16 * S + kc, S, lane);
    v16bf bV2 = ldfrag(Vp + 2L * 16 * S + kc, S, lane);
    v16bf bV3 = ldfrag(Vp + 3L * 16 * S + kc, S, lane);
    o0 = wmma_bf16(aP, bV0, o0);
    o1 = wmma_bf16(aP, bV1, o1);
    o2 = wmma_bf16(aP, bV2, o2);
    o3 = wmma_bf16(aP, bV3, o3);
  }

  // normalize and store attn[b, s, h*64 + d]
#pragma unroll
  for (int j = 0; j < 8; ++j) {
    const int s = q0 + hs * 8 + j;
    const float inv = 1.0f / l_[j];
    const long base = ((long)b * S + s) * E + h * HD;
    attnb[base + 0 * 16 + n] = (__bf16)(o0[j] * inv);
    attnb[base + 1 * 16 + n] = (__bf16)(o1[j] * inv);
    attnb[base + 2 * 16 + n] = (__bf16)(o2[j] * inv);
    attnb[base + 3 * 16 + n] = (__bf16)(o3[j] * inv);
  }
}

// ---------------------------------------------------------------------------
// Output projection + bias + residual:
//   y[b,e,s] = attn[b,s,:] @ Wo[:,e] + bo[e] + x[b,e,s]     (f32, [B][E][S])
// ---------------------------------------------------------------------------
__global__ void __launch_bounds__(256) k_gemm_out(
    const __bf16* __restrict__ attnb, const __bf16* __restrict__ Wot,
    const float* __restrict__ bo, const float* __restrict__ x,
    float* __restrict__ y) {
  using namespace cfg;
  __shared__ __bf16 sA[2][128][32];
  __shared__ __bf16 sB[2][128][32];

  const int b = blockIdx.z;
  const int tid = threadIdx.x;
  const int lane = tid & 31, w = tid >> 5;
  const int sBlk = blockIdx.y * 128, eBlk = blockIdx.x * 128;

  v8f acc[2][4] = {};
  gemm_core_128x128(attnb + ((long)b * S + sBlk) * E, Wot + (long)eBlk * E, E,
                    sA, sB, lane, w, tid, acc);

  const int n = lane & 15, hs = lane >> 4;
  const int s0w = sBlk + (w >> 1) * 32;
  const int e0w = eBlk + (w & 1) * 64;
#pragma unroll
  for (int mi = 0; mi < 2; ++mi) {
#pragma unroll
    for (int ni = 0; ni < 4; ++ni) {
      const int e = e0w + ni * 16 + n;
      const float bb = bo[e];
      const long rowbase = ((long)b * E + e) * S;
#pragma unroll
      for (int j = 0; j < 8; ++j) {
        const int s = s0w + mi * 16 + hs * 8 + j;
        y[rowbase + s] = acc[mi][ni][j] + bb + x[rowbase + s];
      }
    }
  }
}

// ---------------------------------------------------------------------------
// GroupNorm: in [B][E][S] layout each group is 16*1024 contiguous floats.
// ---------------------------------------------------------------------------
__global__ void __launch_bounds__(256) k_gn_stats(
    const float* __restrict__ y, float* __restrict__ stats) {
  using namespace cfg;
  constexpr int N = CPG * S;  // 16384
  const int bg = blockIdx.x;  // b*NG + g
  const float* base = y + (long)bg * N;
  float sum = 0.0f, ss = 0.0f;
  for (int i = threadIdx.x; i < N; i += 256) {
    const float v = base[i];
    sum += v; ss += v * v;
  }
  __shared__ float r0[256], r1[256];
  const int tid = threadIdx.x;
  r0[tid] = sum; r1[tid] = ss;
  __syncthreads();
  for (int off = 128; off > 0; off >>= 1) {
    if (tid < off) { r0[tid] += r0[tid + off]; r1[tid] += r1[tid + off]; }
    __syncthreads();
  }
  if (tid == 0) {
    const float mean = r0[0] * (1.0f / N);
    const float var = r1[0] * (1.0f / N) - mean * mean;
    stats[bg * 2 + 0] = mean;
    stats[bg * 2 + 1] = rsqrtf(var + EPS);
  }
}

__global__ void __launch_bounds__(256) k_gn_apply(
    const float* __restrict__ y, const float* __restrict__ stats,
    const float* __restrict__ gamma, const float* __restrict__ beta,
    float* __restrict__ out) {
  using namespace cfg;
  const long i = (long)blockIdx.x * 256 + threadIdx.x;  // B*E*S total
  const long t = i >> 10;            // S = 1024
  const int e = (int)(t & (E - 1));
  const int b = (int)(t >> 9);       // E = 512
  const int g = e >> 4;              // 16 channels per group
  const float mean = stats[(b * NG + g) * 2 + 0];
  const float rstd = stats[(b * NG + g) * 2 + 1];
  out[i] = (y[i] - mean) * rstd * gamma[e] + beta[e];
}

// ---------------------------------------------------------------------------
// Host launcher
// ---------------------------------------------------------------------------
extern "C" void kernel_launch(void* const* d_in, const int* in_sizes, int n_in,
                              void* d_out, int out_size, void* d_ws, size_t ws_size,
                              hipStream_t stream) {
  using namespace cfg;
  const float* x     = (const float*)d_in[0];  // [B,C,H,W] = [B,C,S]
  const float* Wq    = (const float*)d_in[1];
  const float* bq    = (const float*)d_in[2];
  const float* Wk    = (const float*)d_in[3];
  const float* bk    = (const float*)d_in[4];
  const float* Wv    = (const float*)d_in[5];
  const float* bv    = (const float*)d_in[6];
  const float* Wo    = (const float*)d_in[7];
  const float* bo    = (const float*)d_in[8];
  const float* gamma = (const float*)d_in[9];
  const float* beta  = (const float*)d_in[10];
  float* out = (float*)d_out;

  // workspace layout (all 256B-aligned by construction; ~66 MB total)
  char* p = (char*)d_ws;
  __bf16* Wts  = (__bf16*)p; p += 3L * E * C * sizeof(__bf16);   // Wq^T,Wk^T,Wv^T
  __bf16* Wot  = (__bf16*)p; p += (long)E * E * sizeof(__bf16);  // Wo^T
  __bf16* xb   = (__bf16*)p; p += (long)B * S * C * sizeof(__bf16);
  __bf16* Qb   = (__bf16*)p; p += (long)B * NH * S * HD * sizeof(__bf16);
  __bf16* Kb   = (__bf16*)p; p += (long)B * NH * S * HD * sizeof(__bf16);
  __bf16* Vb   = (__bf16*)p; p += (long)B * NH * S * HD * sizeof(__bf16);
  __bf16* Vtb  = (__bf16*)p; p += (long)B * NH * HD * S * sizeof(__bf16);
  __bf16* attn = (__bf16*)p; p += (long)B * S * E * sizeof(__bf16);
  float*  yb   = (float*)p;  p += (long)B * E * S * sizeof(float);
  float*  st   = (float*)p;  p += (long)B * NG * 2 * sizeof(float);
  (void)ws_size; (void)in_sizes; (void)n_in; (void)out_size;

  // 1) weight transposes: f32 [C][E] -> bf16 [E][C]
  {
    dim3 g(E / 32, C / 32, 1);
    k_transpose_f32_bf16<<<g, 256, 0, stream>>>(Wq, Wts + 0L * E * C, C, E, 0, 0);
    k_transpose_f32_bf16<<<g, 256, 0, stream>>>(Wk, Wts + 1L * E * C, C, E, 0, 0);
    k_transpose_f32_bf16<<<g, 256, 0, stream>>>(Wv, Wts + 2L * E * C, C, E, 0, 0);
    k_transpose_f32_bf16<<<g, 256, 0, stream>>>(Wo, Wot, C, E, 0, 0);
  }
  // 2) x: f32 [B][C][S] -> bf16 [B][S][C]
  {
    dim3 g(S / 32, C / 32, B);
    k_transpose_f32_bf16<<<g, 256, 0, stream>>>(x, xb, C, S,
                                                (long)C * S, (long)S * C);
  }
  // 3) QKV projection GEMMs (WMMA + TDM/async LDS staging)
  {
    dim3 g(E / 128, S / 128, B * 3);
    k_gemm_qkv<<<g, 256, 0, stream>>>(xb, Wts, bq, bk, bv, Qb, Kb, Vb);
  }
  // 4) V -> V^T per head: bf16 [S][HD] -> [HD][S]
  {
    dim3 g(HD / 32, S / 32, B * NH);
    k_transpose_bf16<<<g, 256, 0, stream>>>(Vb, Vtb, S, HD,
                                            (long)S * HD, (long)HD * S);
  }
  // 5) flash attention (WMMA)
  {
    dim3 g(S / 128, B * NH, 1);
    k_attention<<<g, 256, 0, stream>>>(Qb, Kb, Vtb, attn);
  }
  // 6) output projection + bias + residual (WMMA + TDM/async LDS staging)
  {
    dim3 g(E / 128, S / 128, B);
    k_gemm_out<<<g, 256, 0, stream>>>(attn, Wot, bo, x, yb);
  }
  // 7) GroupNorm
  k_gn_stats<<<dim3(B * NG), 256, 0, stream>>>(yb, st);
  k_gn_apply<<<dim3((B * E * S) / 256), 256, 0, stream>>>(yb, st, gamma, beta, out);
}
